// MultiHeadAttention_12730283065384
// MI455X (gfx1250) — matrix operand order repaired and checked by reading, not compiled, not run on hardware
//
#include <hip/hip_runtime.h>
#include <stdint.h>

#define H_HEADS   16
#define DH        64
#define LQ_LEN    1024
#define BATCH     4
#define DMODEL    1024
#define LIMIT_RHS 4

typedef __attribute__((ext_vector_type(16))) _Float16 v16h;
typedef __attribute__((ext_vector_type(8)))  float    v8f;
typedef __attribute__((ext_vector_type(4)))  int      v4i;

union Frag {
  v16h h;
  uint32_t u[8];
};

__device__ __forceinline__ v8f wmma_f32_f16(v16h a, v16h b, v8f c) {
  // D = A(16x32 f16) * B(32x16 f16) + C(16x16 f32)
  return __builtin_amdgcn_wmma_f32_16x16x32_f16(false, a, false, b, (short)0, c,
                                                false, false);
}

// ---------------------------------------------------------------------------
// Async global->LDS (CDNA5 GLOBAL_LOAD_ASYNC_TO_LDS_B128), guarded so the
// build falls back to synchronous staging if the builtins are absent.
// Probe-derived parameter type: int4 in AS1 (global), int4 in AS3 (LDS).
// ---------------------------------------------------------------------------
#if __has_builtin(__builtin_amdgcn_global_load_async_to_lds_b128) && \
    __has_builtin(__builtin_amdgcn_s_wait_asynccnt)
#define HAVE_ASYNC_LDS 1
#else
#define HAVE_ASYNC_LDS 0
#endif

typedef __attribute__((address_space(1))) v4i* gv4i_p;
typedef __attribute__((address_space(3))) v4i* lv4i_p;

__device__ __forceinline__ void copy16_g2l(const _Float16* g, _Float16* l) {
#if HAVE_ASYNC_LDS
  __builtin_amdgcn_global_load_async_to_lds_b128(
      (gv4i_p)const_cast<_Float16*>(g), (lv4i_p)l, 0, 0);
#else
  *(uint4*)l = *(const uint4*)g;
#endif
}

__device__ __forceinline__ void wait_async_then_barrier() {
#if HAVE_ASYNC_LDS
  __builtin_amdgcn_s_wait_asynccnt(0);
#endif
  __syncthreads();
}

// ---------------------------------------------------------------------------
// f32 -> f16 conversion / zero init
// ---------------------------------------------------------------------------
__global__ void convert_f32_to_f16(const float* __restrict__ in,
                                   _Float16* __restrict__ out, int n) {
  int i = blockIdx.x * blockDim.x + threadIdx.x;
  if (i < n) out[i] = (_Float16)in[i];
}

__global__ void zero_f32(float* __restrict__ p, int n) {
  int i = blockIdx.x * blockDim.x + threadIdx.x;
  if (i < n) p[i] = 0.0f;
}

// ---------------------------------------------------------------------------
// C = A(MxK) * B(NxK)^T  with WMMA f16->f32, async double-buffered LDS.
// mode 0: f32 row-major (M x N) output
// mode 1: f16 head layout   dst[b][h][lq][dh]   (row = lq*4+b, col = h*64+dh)
// mode 2: f16 head-T layout dst[b][h][dh][lq]   (V, so P@V B-frags are contiguous)
// ---------------------------------------------------------------------------
#define TM 128
#define TN 64
#define TK 64

__global__ __launch_bounds__(256) void gemm_bt(const _Float16* __restrict__ A,
                                               const _Float16* __restrict__ Bw,
                                               void* __restrict__ Cout,
                                               int M, int N, int K,
                                               int mode, float scale) {
  __shared__ _Float16 As[2 * TM * TK];   // 32 KB
  __shared__ _Float16 Bs[2 * TN * TK];   // 16 KB

  const int tid  = threadIdx.x;
  const int lane = tid & 31;
  const int wave = tid >> 5;      // 0..7
  const int wm   = wave & 3;      // 4 waves along M
  const int wn   = wave >> 2;     // 2 waves along N
  const int l16  = lane & 15;
  const int half = lane >> 4;
  const int m0   = blockIdx.x * TM;
  const int n0   = blockIdx.y * TN;

  // staging assignment: A slab 128x64 f16 -> 64B per thread,
  //                     B slab  64x64 f16 -> 32B per thread
  const int arow = tid >> 1;           // 0..127
  const int acol = (tid & 1) * 32;     // f16 column base
  const int brow = tid >> 2;           // 0..63
  const int bcol = (tid & 3) * 16;

  v8f acc[2][2];
#pragma unroll
  for (int mt = 0; mt < 2; ++mt)
#pragma unroll
    for (int nt = 0; nt < 2; ++nt)
#pragma unroll
      for (int r = 0; r < 8; ++r) acc[mt][nt][r] = 0.0f;

  const int nk = K / TK;

  // prologue: stage slab 0 into buffer 0
  {
    _Float16* asl = As + arow * TK + acol;
    const _Float16* ag = A + (size_t)(m0 + arow) * K + acol;
    copy16_g2l(ag, asl);
    copy16_g2l(ag + 8,  asl + 8);
    copy16_g2l(ag + 16, asl + 16);
    copy16_g2l(ag + 24, asl + 24);
    _Float16* bsl = Bs + brow * TK + bcol;
    const _Float16* bg = Bw + (size_t)(n0 + brow) * K + bcol;
    copy16_g2l(bg, bsl);
    copy16_g2l(bg + 8, bsl + 8);
  }

  for (int kt = 0; kt < nk; ++kt) {
    wait_async_then_barrier();   // slab kt resident in LDS, all waves synced

    if (kt + 1 < nk) {           // prefetch slab kt+1 into the other buffer
      const int k0   = (kt + 1) * TK;
      const int bsel = (kt + 1) & 1;
      _Float16* asl = As + bsel * TM * TK + arow * TK + acol;
      const _Float16* ag = A + (size_t)(m0 + arow) * K + k0 + acol;
      copy16_g2l(ag, asl);
      copy16_g2l(ag + 8,  asl + 8);
      copy16_g2l(ag + 16, asl + 16);
      copy16_g2l(ag + 24, asl + 24);
      _Float16* bsl = Bs + bsel * TN * TK + brow * TK + bcol;
      const _Float16* bg = Bw + (size_t)(n0 + brow) * K + k0 + bcol;
      copy16_g2l(bg, bsl);
      copy16_g2l(bg + 8, bsl + 8);
    }

    const _Float16* Ab = As + (kt & 1) * TM * TK;
    const _Float16* Bb = Bs + (kt & 1) * TN * TK;

#pragma unroll
    for (int kc = 0; kc < 2; ++kc) {   // two K=32 chunks per slab
      Frag a[2], b[2];
#pragma unroll
      for (int mt = 0; mt < 2; ++mt) {
        const _Float16* ap = Ab + (size_t)(wm * 32 + mt * 16 + l16) * TK + kc * 32;
#pragma unroll
        for (int v = 0; v < 8; ++v)
          a[mt].u[v] =
              *(const uint32_t*)(ap + (half * 16 + v * 2));
      }
#pragma unroll
      for (int nt = 0; nt < 2; ++nt) {
        const _Float16* bp = Bb + (size_t)(wn * 32 + nt * 16 + l16) * TK + kc * 32;
#pragma unroll
        for (int v = 0; v < 8; ++v)
          b[nt].u[v] = *(const uint32_t*)(bp + (half * 16 + v * 2));
      }
#pragma unroll
      for (int mt = 0; mt < 2; ++mt)
#pragma unroll
        for (int nt = 0; nt < 2; ++nt)
          acc[mt][nt] = wmma_f32_f16(a[mt].h, b[nt].h, acc[mt][nt]);
    }
  }

#pragma unroll
  for (int mt = 0; mt < 2; ++mt)
#pragma unroll
    for (int nt = 0; nt < 2; ++nt) {
      const int col = n0 + wn * 32 + nt * 16 + l16;
#pragma unroll
      for (int r = 0; r < 8; ++r) {
        const int rowi = m0 + wm * 32 + mt * 16 + r + 8 * half;
        const float val = acc[mt][nt][r] * scale;
        if (mode == 0) {
          ((float*)Cout)[(size_t)rowi * N + col] = val;
        } else {
          const int lq  = rowi >> 2;   // row = lq*BATCH + b
          const int bb  = rowi & 3;
          const int hh  = col >> 6;    // col = h*DH + dh
          const int dhi = col & 63;
          if (mode == 1)
            ((_Float16*)Cout)[(((size_t)bb * H_HEADS + hh) * LQ_LEN + lq) * DH + dhi] =
                (_Float16)val;
          else
            ((_Float16*)Cout)[(((size_t)bb * H_HEADS + hh) * DH + dhi) * LQ_LEN + lq] =
                (_Float16)val;
        }
      }
    }
}

// ---------------------------------------------------------------------------
// Causal attention, one wave per (b, h, 16-row q tile). Two passes:
//   pass 1: streaming row max + row sum (online softmax stats)
//   pass 2: recompute scores, p = exp(s-m)/sum, coverage += p/H,
//           P (C-layout) -> LDS -> A-layout, O += P@V via WMMA
// ---------------------------------------------------------------------------
__global__ __launch_bounds__(32) void attn_kernel(const _Float16* __restrict__ Qs,
                                                  const _Float16* __restrict__ Ks,
                                                  const _Float16* __restrict__ Vt,
                                                  _Float16* __restrict__ AO,
                                                  float* __restrict__ cov,
                                                  const int* __restrict__ layer_idx_p) {
  __shared__ _Float16 Pt[16 * 32];

  const int lane = threadIdx.x & 31;
  const int l16  = lane & 15;
  const int half = lane >> 4;

  const int bid = blockIdx.x;          // b*1024 + h*64 + qt
  const int qt  = bid & 63;
  const int h   = (bid >> 6) & 15;
  const int b   = bid >> 10;
  const int q0  = qt * 16;

  // rhs mask threshold (for layer_idx=2 this reduces to pure causal)
  const int layer_idx = *layer_idx_p;
  int off = layer_idx * LIMIT_RHS;
  int t0  = LIMIT_RHS - off; if (t0 < 0) t0 = 0;
  const int start = LIMIT_RHS - t0;
  const int delta = LIMIT_RHS - start;     // rhs allows j <= i + delta
  const int dmin  = delta < 0 ? delta : 0; // combined with causal: j <= i + dmin

  const _Float16* qb = Qs + (((size_t)b * H_HEADS + h) * LQ_LEN + q0) * DH;
  const _Float16* kb = Ks + (((size_t)b * H_HEADS + h) * LQ_LEN) * DH;
  const _Float16* vb = Vt + (((size_t)b * H_HEADS + h) * DH) * LQ_LEN;

  // Q A-fragments (16 x 64 split into two 16x32)
  Frag aq[2];
#pragma unroll
  for (int f = 0; f < 2; ++f)
#pragma unroll
    for (int v = 0; v < 8; ++v) {
      const int kk = f * 32 + half * 16 + v * 2;
      aq[f].u[v] = *(const uint32_t*)(qb + (size_t)l16 * DH + kk);
    }

  const int ktmax = (q0 + 15) >> 4;  // inclusive last 16-wide k tile (causal)

  float rmax[8], rsum[8];
#pragma unroll
  for (int r = 0; r < 8; ++r) { rmax[r] = -__builtin_inff(); rsum[r] = 0.0f; }

  // ---------------- pass 1: softmax statistics ----------------
  for (int kt = 0; kt <= ktmax; ++kt) {
    const int k0 = kt * 16;
    Frag bk[2];
#pragma unroll
    for (int f = 0; f < 2; ++f)
#pragma unroll
      for (int v = 0; v < 8; ++v) {
        const int kk = f * 32 + half * 16 + v * 2;
        bk[f].u[v] = *(const uint32_t*)(kb + (size_t)(k0 + l16) * DH + kk);
      }
    v8f c;
#pragma unroll
    for (int r = 0; r < 8; ++r) c[r] = 0.0f;
    c = wmma_f32_f16(aq[0].h, bk[0].h, c);
    c = wmma_f32_f16(aq[1].h, bk[1].h, c);

    const int j = k0 + l16;
#pragma unroll
    for (int r = 0; r < 8; ++r) {
      const int i = q0 + r + 8 * half;
      float s = (j <= i + dmin) ? c[r] : -__builtin_inff();
      float mx = s;
#pragma unroll
      for (int o = 1; o < 16; o <<= 1) mx = fmaxf(mx, __shfl_xor(mx, o, 32));
      const float mnew = fmaxf(rmax[r], mx);
      const float corr = __expf(rmax[r] - mnew);
      float e = __expf(s - mnew);
#pragma unroll
      for (int o = 1; o < 16; o <<= 1) e += __shfl_xor(e, o, 32);
      rsum[r] = rsum[r] * corr + e;
      rmax[r] = mnew;
    }
  }

  float inv[8];
#pragma unroll
  for (int r = 0; r < 8; ++r) inv[r] = 1.0f / rsum[r];

  // ---------------- pass 2: probabilities, coverage, O = P@V ----------------
  v8f co[4];
#pragma unroll
  for (int nt = 0; nt < 4; ++nt)
#pragma unroll
    for (int r = 0; r < 8; ++r) co[nt][r] = 0.0f;

  const int npair = (ktmax + 2) >> 1;  // 32-wide K pairs (zero-pad odd tail)
  for (int tp = 0; tp < npair; ++tp) {
#pragma unroll
    for (int sub = 0; sub < 2; ++sub) {
      const int kt = tp * 2 + sub;
      if (kt <= ktmax) {
        const int k0 = kt * 16;
        Frag bk[2];
#pragma unroll
        for (int f = 0; f < 2; ++f)
#pragma unroll
          for (int v = 0; v < 8; ++v) {
            const int kk = f * 32 + half * 16 + v * 2;
            bk[f].u[v] = *(const uint32_t*)(kb + (size_t)(k0 + l16) * DH + kk);
          }
        v8f c;
#pragma unroll
        for (int r = 0; r < 8; ++r) c[r] = 0.0f;
        c = wmma_f32_f16(aq[0].h, bk[0].h, c);
        c = wmma_f32_f16(aq[1].h, bk[1].h, c);

        const int j = k0 + l16;
#pragma unroll
        for (int r = 0; r < 8; ++r) {
          const int i  = q0 + r + 8 * half;
          const bool ok = (j <= i + dmin);
          const float s = ok ? c[r] : -__builtin_inff();
          const float p = __expf(s - rmax[r]) * inv[r];
          Pt[(r + 8 * half) * 32 + sub * 16 + l16] = (_Float16)p;
          if (ok)
            atomicAdd(cov + ((size_t)b * LQ_LEN + i) * LQ_LEN + j,
                      p * (1.0f / H_HEADS));
        }
      } else {
#pragma unroll
        for (int r = 0; r < 8; ++r)
          Pt[(r + 8 * half) * 32 + sub * 16 + l16] = (_Float16)0.0f;
      }
    }
    __syncthreads();

    // reload P in A-fragment layout
    Frag pa;
#pragma unroll
    for (int v = 0; v < 8; ++v) {
      const int kk = half * 16 + v * 2;
      pa.u[v] = *(const uint32_t*)(Pt + (size_t)l16 * 32 + kk);
    }

    const int kbase = tp * 32;
#pragma unroll
    for (int nt = 0; nt < 4; ++nt) {
      Frag bv;  // V^T stored [dh][lq] -> contiguous K per lane
#pragma unroll
      for (int v = 0; v < 8; ++v) {
        const int kk = half * 16 + v * 2;
        bv.u[v] = *(const uint32_t*)(vb + (size_t)(nt * 16 + l16) * LQ_LEN + kbase + kk);
      }
      co[nt] = wmma_f32_f16(pa.h, bv.h, co[nt]);
    }
    __syncthreads();
  }

  // write O in (lq, b, d) f16 layout for the Wo GEMM
#pragma unroll
  for (int nt = 0; nt < 4; ++nt)
#pragma unroll
    for (int r = 0; r < 8; ++r) {
      const int i = q0 + r + 8 * half;
      AO[((size_t)i * BATCH + b) * DMODEL + h * DH + nt * 16 + l16] =
          (_Float16)co[nt][r];
    }
}

// ---------------------------------------------------------------------------
extern "C" void kernel_launch(void* const* d_in, const int* in_sizes, int n_in,
                              void* d_out, int out_size, void* d_ws, size_t ws_size,
                              hipStream_t stream) {
  (void)in_sizes; (void)n_in; (void)out_size; (void)ws_size;

  const float* query = (const float*)d_in[0];
  const float* keyt  = (const float*)d_in[1];
  const float* value = (const float*)d_in[2];
  const float* Wq    = (const float*)d_in[3];
  const float* Wk    = (const float*)d_in[4];
  const float* Wv    = (const float*)d_in[5];
  const float* Wo    = (const float*)d_in[6];
  const int*   layer = (const int*)d_in[9];

  _Float16* ws = (_Float16*)d_ws;
  const size_t SX = (size_t)4096 * 1024;  // LQ*B x D
  const size_t SW = (size_t)1024 * 1024;  // D x D
  _Float16* Xq   = ws;
  _Float16* Xk   = Xq + SX;
  _Float16* Xv   = Xk + SX;
  _Float16* Wq16 = Xv + SX;
  _Float16* Wk16 = Wq16 + SW;
  _Float16* Wv16 = Wk16 + SW;
  _Float16* Wo16 = Wv16 + SW;
  _Float16* Qs   = Wo16 + SW;
  _Float16* Ks   = Qs + SX;
  _Float16* Vt   = Ks + SX;
  _Float16* AO   = Vt + SX;   // total 64 MB f16 workspace

  const int nX = 4096 * 1024, nW = 1024 * 1024;
  convert_f32_to_f16<<<(nX + 255) / 256, 256, 0, stream>>>(query, Xq, nX);
  convert_f32_to_f16<<<(nX + 255) / 256, 256, 0, stream>>>(keyt,  Xk, nX);
  convert_f32_to_f16<<<(nX + 255) / 256, 256, 0, stream>>>(value, Xv, nX);
  convert_f32_to_f16<<<(nW + 255) / 256, 256, 0, stream>>>(Wq, Wq16, nW);
  convert_f32_to_f16<<<(nW + 255) / 256, 256, 0, stream>>>(Wk, Wk16, nW);
  convert_f32_to_f16<<<(nW + 255) / 256, 256, 0, stream>>>(Wv, Wv16, nW);
  convert_f32_to_f16<<<(nW + 255) / 256, 256, 0, stream>>>(Wo, Wo16, nW);

  dim3 ggrid(4096 / TM, 1024 / TN);
  // Q scaled by dh^-0.5 = 0.125 in the projection epilogue
  gemm_bt<<<ggrid, 256, 0, stream>>>(Xq, Wq16, Qs, 4096, 1024, 1024, 1, 0.125f);
  gemm_bt<<<ggrid, 256, 0, stream>>>(Xk, Wk16, Ks, 4096, 1024, 1024, 1, 1.0f);
  gemm_bt<<<ggrid, 256, 0, stream>>>(Xv, Wv16, Vt, 4096, 1024, 1024, 2, 1.0f);

  float* out = (float*)d_out;
  float* cov = out + (size_t)4096 * 1024;
  zero_f32<<<(4194304 + 255) / 256, 256, 0, stream>>>(cov, 4194304);

  attn_kernel<<<BATCH * H_HEADS * (LQ_LEN / 16), 32, 0, stream>>>(
      Qs, Ks, Vt, AO, cov, layer);

  gemm_bt<<<ggrid, 256, 0, stream>>>(AO, Wo16, out, 4096, 1024, 1024, 0, 1.0f);
}
